// PoolOnlyGNN_32212254720737
// MI455X (gfx1250) — matrix operand order repaired and checked by reading, not compiled, not run on hardware
//
#include <hip/hip_runtime.h>

#define D 128
#define NEG_SLOPE 0.01f

typedef __attribute__((ext_vector_type(16))) __bf16 v16bf;
typedef __attribute__((ext_vector_type(8)))  __bf16 v8bf;
typedef __attribute__((ext_vector_type(8)))  float  v8f;

__device__ __forceinline__ unsigned short f2bf(float f) {
  unsigned int u = __float_as_uint(f);
  u += 0x7FFFu + ((u >> 16) & 1u);          // round-to-nearest-even
  return (unsigned short)(u >> 16);
}
__device__ __forceinline__ float leaky(float v) { return v >= 0.f ? v : NEG_SLOPE * v; }

// ---------------- copy x -> out (x is returned unchanged) ----------------
__global__ __launch_bounds__(256) void k_copy_x(const float4* __restrict__ x,
                                                float4* __restrict__ out, int n4) {
  int i = blockIdx.x * 256 + threadIdx.x;
  if (i < n4) out[i] = x[i];
}

// ---------------- CSR segment offsets from sorted int64 batch_ind ----------------
__global__ __launch_bounds__(256) void k_segoff(const long long* __restrict__ b,
                                                int N, int G, int* __restrict__ off) {
  int n = blockIdx.x * 256 + threadIdx.x;
  if (n > N) return;
  int lo = (n == 0) ? 0 : (int)b[n - 1] + 1;
  int hi = (n == N) ? G : (int)b[n];
  for (int g = lo; g <= hi; ++g) off[g] = n;   // off[g] = first n with b[n] >= g
}

// ---------------- gate[n] = x[n] . gate_w + gate_b (one wave per node) ----------------
__global__ __launch_bounds__(256) void k_gate(const float* __restrict__ x,
                                              const float* __restrict__ gw,
                                              const float* __restrict__ gb,
                                              float* __restrict__ gate, int N) {
  int node = (blockIdx.x * 256 + threadIdx.x) >> 5;
  int lane = threadIdx.x & 31;
  if (node >= N) return;
  float4 xv = ((const float4*)(x + (size_t)node * D))[lane];
  float4 wv = ((const float4*)gw)[lane];
  float s = xv.x * wv.x + xv.y * wv.y + xv.z * wv.z + xv.w * wv.w;
  #pragma unroll
  for (int m = 16; m; m >>= 1) s += __shfl_xor(s, m, 32);
  if (lane == 0) gate[node] = s + gb[0];
}

// ---------------- feat_w (f32, [k][n]) -> bf16 transposed [n][k] ----------------
__global__ __launch_bounds__(256) void k_convw(const float* __restrict__ fw,
                                               unsigned short* __restrict__ bw) {
  int idx = blockIdx.x * 256 + threadIdx.x;
  if (idx >= D * D) return;
  int n = idx >> 7, k = idx & 127;
  bw[idx] = f2bf(fw[k * D + n]);
}

// ---------------- per-graph: softmax stats + WMMA feat matmul + weighted pool ----------------
#define BSTRIDE 136   // padded LDS row stride (bf16 elems): 272B -> rows land on different banks
__global__ __launch_bounds__(256) void k_pool(
    const float* __restrict__ x, const float* __restrict__ gate,
    const int* __restrict__ segoff, const unsigned short* __restrict__ bw,
    const float* __restrict__ fb, float* __restrict__ pooled) {
  __shared__ unsigned short ldsB[D * BSTRIDE];
  __shared__ unsigned short ldsA[2][16 * BSTRIDE];   // double-buffered A tiles
  __shared__ float alphaS[2][16];
  __shared__ float red[8];
  __shared__ float stat[2];

  const int g    = blockIdx.x;
  const int tid  = threadIdx.x;
  const int lane = tid & 31;
  const int wid  = tid >> 5;
  const int h    = lane >> 4;        // lane half (wave32)
  const int l16  = lane & 15;
  const int ncol = (wid << 4) | l16; // this lane's output column (wave owns 16 cols)

  // stage transposed bf16 weights into LDS (coalesced uint4 loads)
  const uint4* bw4 = (const uint4*)bw;           // 2048 x 16B
  #pragma unroll
  for (int i = 0; i < 8; ++i) {
    int idx = tid + i * 256;
    int e0 = idx << 3;
    int n = e0 >> 7, k = e0 & 127;
    *(uint4*)&ldsB[n * BSTRIDE + k] = bw4[idx];
  }
  __syncthreads();

  // B fragments resident in VGPRs for the whole block: b[e] = Bt[ncol][32c + 16h + e]
  v16bf bfrag[4];
  #pragma unroll
  for (int c = 0; c < 4; ++c) {
    int base = ncol * BSTRIDE + 32 * c + 16 * h;
    v8bf lo = *(const v8bf*)&ldsB[base];
    v8bf hi = *(const v8bf*)&ldsB[base + 8];
    bfrag[c] = __builtin_shufflevector(lo, hi, 0,1,2,3,4,5,6,7,8,9,10,11,12,13,14,15);
  }
  const float fbn = fb[ncol];

  const int s0 = segoff[g], s1 = segoff[g + 1];

  // segment softmax: max
  float lm = -3.4e38f;
  for (int n = s0 + tid; n < s1; n += 256) lm = fmaxf(lm, gate[n]);
  #pragma unroll
  for (int m = 16; m; m >>= 1) lm = fmaxf(lm, __shfl_xor(lm, m, 32));
  if (lane == 0) red[wid] = lm;
  __syncthreads();
  if (tid == 0) {
    float mm = red[0];
    for (int j = 1; j < 8; ++j) mm = fmaxf(mm, red[j]);
    stat[0] = mm;
  }
  __syncthreads();
  const float mx = stat[0];
  // segment softmax: denominator
  float ls = 0.f;
  for (int n = s0 + tid; n < s1; n += 256) ls += __expf(gate[n] - mx);
  #pragma unroll
  for (int m = 16; m; m >>= 1) ls += __shfl_xor(ls, m, 32);
  if (lane == 0) red[wid] = ls;
  __syncthreads();
  if (tid == 0) {
    float ss = red[0];
    for (int j = 1; j < 8; ++j) ss += red[j];
    stat[1] = ss;
  }
  __syncthreads();
  const float rden = (stat[1] > 0.f) ? (1.f / stat[1]) : 0.f;

  const int arow = tid >> 4;          // A staging: row 0..15
  const int acb  = (tid & 15) << 3;   // A staging: col base 0..120 step 8
  const int nt   = (s1 - s0 + 15) >> 4;

  // prologue: stage tile 0 into buffer 0
  if (nt > 0) {
    int node = s0 + arow;
    uint4 pk = {0u, 0u, 0u, 0u};
    if (node < s1) {
      const float4* xp = (const float4*)(x + (size_t)node * D + acb);
      float4 v0 = xp[0], v1 = xp[1];
      pk.x = (unsigned)f2bf(v0.x) | ((unsigned)f2bf(v0.y) << 16);
      pk.y = (unsigned)f2bf(v0.z) | ((unsigned)f2bf(v0.w) << 16);
      pk.z = (unsigned)f2bf(v1.x) | ((unsigned)f2bf(v1.y) << 16);
      pk.w = (unsigned)f2bf(v1.z) | ((unsigned)f2bf(v1.w) << 16);
    }
    *(uint4*)&ldsA[0][arow * BSTRIDE + acb] = pk;
    if (tid < 16) {
      int n = s0 + tid;
      alphaS[0][tid] = (n < s1) ? __expf(gate[n] - mx) * rden : 0.f;
    }
  }
  __syncthreads();

  v8f acc = {0.f, 0.f, 0.f, 0.f, 0.f, 0.f, 0.f, 0.f};

  for (int t = 0; t < nt; ++t) {
    const int cur = t & 1, nxt = cur ^ 1;
    const bool have_next = (t + 1 < nt);

    // fetch tile t+1 into registers (overlaps with WMMA below)
    uint4 pk = {0u, 0u, 0u, 0u};
    float anew = 0.f;
    if (have_next) {
      int node = s0 + (t + 1) * 16 + arow;
      if (node < s1) {
        const float4* xp = (const float4*)(x + (size_t)node * D + acb);
        float4 v0 = xp[0], v1 = xp[1];
        pk.x = (unsigned)f2bf(v0.x) | ((unsigned)f2bf(v0.y) << 16);
        pk.y = (unsigned)f2bf(v0.z) | ((unsigned)f2bf(v0.w) << 16);
        pk.z = (unsigned)f2bf(v1.x) | ((unsigned)f2bf(v1.y) << 16);
        pk.w = (unsigned)f2bf(v1.z) | ((unsigned)f2bf(v1.w) << 16);
      }
      if (tid < 16) {
        int n = s0 + (t + 1) * 16 + tid;
        anew = (n < s1) ? __expf(gate[n] - mx) * rden : 0.f;
      }
      // prefetch tile t+2 (global_prefetch_b8)
      int node2 = s0 + (t + 2) * 16 + arow;
      if (t + 2 < nt && node2 < s1)
        __builtin_prefetch(x + (size_t)node2 * D + acb, 0, 0);
    }

    // compute tile t: D = A(16x128 bf16) x B(128x16 bf16), K chunked by 32
    v8f d = {0.f, 0.f, 0.f, 0.f, 0.f, 0.f, 0.f, 0.f};
    #pragma unroll
    for (int c = 0; c < 4; ++c) {
      // a[e] = A[M][32c+16h+e]  (M = lane%16)
      int base = l16 * BSTRIDE + 32 * c + 16 * h;
      v8bf lo = *(const v8bf*)&ldsA[cur][base];
      v8bf hi = *(const v8bf*)&ldsA[cur][base + 8];
      v16bf a = __builtin_shufflevector(lo, hi, 0,1,2,3,4,5,6,7,8,9,10,11,12,13,14,15);
      d = __builtin_amdgcn_wmma_f32_16x16x32_bf16(false, a, false, bfrag[c],
                                                  (short)0, d, false, false);
    }
    // bias + leaky + alpha-weighted accumulate (row M = v + 8h)
    #pragma unroll
    for (int v = 0; v < 8; ++v) {
      float val = leaky(d[v] + fbn);
      acc[v] += alphaS[cur][v + 8 * h] * val;
    }

    // publish tile t+1 into the alternate buffer
    if (have_next) {
      *(uint4*)&ldsA[nxt][arow * BSTRIDE + acb] = pk;
      if (tid < 16) alphaS[nxt][tid] = anew;
    }
    __syncthreads();
  }

  // reduce rows: sum own 8 rows, then add the other lane-half (same column)
  float cs = 0.f;
  #pragma unroll
  for (int v = 0; v < 8; ++v) cs += acc[v];
  cs += __shfl_xor(cs, 16, 32);
  if (h == 0) pooled[g * D + ncol] = cs;
}

// ---------------- xg = leaky([pooled, xg] @ tr_w + tr_b) + xg (fp32, tiny) ----------------
__global__ __launch_bounds__(128) void k_transform(
    const float* __restrict__ pooled, const float* __restrict__ xg_in,
    const float* __restrict__ tw, const float* __restrict__ tb,
    float* __restrict__ xg_out) {
  __shared__ float cat[2 * D];
  int g = blockIdx.x, t = threadIdx.x;
  cat[t] = pooled[g * D + t];
  cat[D + t] = xg_in[g * D + t];
  __syncthreads();
  float s = tb[t];
  #pragma unroll 8
  for (int k = 0; k < 2 * D; ++k) s += cat[k] * tw[k * D + t];
  xg_out[g * D + t] = leaky(s) + cat[D + t];
}

extern "C" void kernel_launch(void* const* d_in, const int* in_sizes, int n_in,
                              void* d_out, int out_size, void* d_ws, size_t ws_size,
                              hipStream_t stream) {
  const float*     x        = (const float*)d_in[0];
  const float*     x_global = (const float*)d_in[1];
  const long long* batch    = (const long long*)d_in[4];   // int64
  const float*     gate_w   = (const float*)d_in[6];
  const float*     gate_b   = (const float*)d_in[7];
  const float*     feat_w   = (const float*)d_in[8];
  const float*     feat_b   = (const float*)d_in[9];
  const float*     tr_w     = (const float*)d_in[10];
  const float*     tr_b     = (const float*)d_in[11];

  const int N = in_sizes[0] / D;   // 1,000,000
  const int G = in_sizes[1] / D;   // 4096
  float* out = (float*)d_out;

  // workspace carve-out (~10.3 MB)
  char* w = (char*)d_ws;
  auto take = [&](size_t bytes) -> char* {
    char* p = w; w += (bytes + 255) & ~(size_t)255; return p;
  };
  float*          gate    = (float*)take((size_t)N * 4);
  float*          pooled  = (float*)take((size_t)G * D * 4);
  unsigned short* bw      = (unsigned short*)take((size_t)D * D * 2);
  int*            segoffp = (int*)take((size_t)(G + 1) * 4);
  float*          xgbuf0  = (float*)take((size_t)G * D * 4);
  float*          xgbuf1  = (float*)take((size_t)G * D * 4);

  {   // x passes through unchanged to output
    int n4 = N * (D / 4);
    k_copy_x<<<(n4 + 255) / 256, 256, 0, stream>>>((const float4*)x, (float4*)out, n4);
  }
  k_segoff<<<(N + 1 + 255) / 256, 256, 0, stream>>>(batch, N, G, segoffp);

  const float* xg_in = x_global;
  for (int i = 0; i < 3; ++i) {
    k_gate<<<(N + 7) / 8, 256, 0, stream>>>(x, gate_w + i * D, gate_b + i, gate, N);
    k_convw<<<(D * D + 255) / 256, 256, 0, stream>>>(feat_w + (size_t)i * D * D, bw);
    k_pool<<<G, 256, 0, stream>>>(x, gate, segoffp, bw, feat_b + i * D, pooled);
    float* xg_out = (i == 2) ? (out + (size_t)N * D) : (i == 0 ? xgbuf0 : xgbuf1);
    k_transform<<<G, 128, 0, stream>>>(pooled, xg_in, tr_w + (size_t)i * 2 * D * D,
                                       tr_b + i * D, xg_out);
    xg_in = xg_out;
  }
  (void)n_in; (void)out_size; (void)ws_size;
}
